// FeatureExtractor_62277025792133
// MI455X (gfx1250) — matrix operand order repaired and checked by reading, not compiled, not run on hardware
//
#include <hip/hip_runtime.h>
#include <hip/hip_bf16.h>

// Problem constants
#define L_  4
#define B_  8
#define S_  2048
#define H_  1024
#define NH_ 2
#define NS_ 32
#define DH_ 512
#define BS_ (B_*S_)

typedef __attribute__((ext_vector_type(16))) _Float16 v16h;
typedef __attribute__((ext_vector_type(8)))  _Float16 v8h;
typedef __attribute__((ext_vector_type(4)))  _Float16 v4h;
typedef __attribute__((ext_vector_type(8)))  float    v8f;

// ---- CDNA5 async global->LDS copy (per-lane b128), tracked by ASYNCcnt ----
__device__ __forceinline__ void async_copy_b128(unsigned lds_off, const void* gptr) {
  asm volatile("global_load_async_to_lds_b128 %0, %1, off"
               :: "v"(lds_off), "v"(gptr) : "memory");
}
__device__ __forceinline__ void wait_async0() {
  asm volatile("s_wait_asynccnt 0x0" ::: "memory");
}

// ---- WMMA fragment loaders (CDNA5 v_wmma_f32_16x16x32_f16 layouts) ----
__device__ __forceinline__ v16h make_frag(v8h lo, v8h hi) {
  v16h r;
#pragma unroll
  for (int i = 0; i < 8; ++i) { r[i] = lo[i]; r[8 + i] = hi[i]; }
  return r;
}

// A matrix 16x32 f16: lane m=lane&15; lanes<16 hold K={0..7,16..23}, lanes>=16 K={8..15,24..31}
__device__ __forceinline__ v16h load_a_frag(const _Float16* row, int k0, int lane) {
  int kh = (lane >> 4) << 3;                 // 0 or 8
  v8h lo = *(const v8h*)(row + k0 + kh);
  v8h hi = *(const v8h*)(row + k0 + kh + 16);
  return make_frag(lo, hi);
}

// B matrix 32x16 f16 (column-major source): lane n=lane&15; lanes<16 K={0..7,16..23}, lanes>=16 K={8..15,24..31}
__device__ __forceinline__ v16h load_b_frag(const _Float16* col, int k0, int lane) {
  int kb = (lane >> 4) << 3;                 // 0 or 8
  v8h lo = *(const v8h*)(col + k0 + kb);
  v8h hi = *(const v8h*)(col + k0 + kb + 16);
  return make_frag(lo, hi);
}

#define WMMA_F16(a, b, c) \
  __builtin_amdgcn_wmma_f32_16x16x32_f16(false, (a), false, (b), (short)0, (c), false, false)

// ---- 1) layer-softmax fusion + LN1 -> x_f32 (residual) and x_f16 (GEMM A) ----
__global__ void __launch_bounds__(256) k_fuse_ln1(
    const float* __restrict__ hs, const float* __restrict__ lw,
    const float* __restrict__ g, const float* __restrict__ bb,
    float* __restrict__ xf32, _Float16* __restrict__ xf16) {
  __shared__ float rs[256], rq[256];
  int row = blockIdx.x;                      // 0..BS-1
  int t = threadIdx.x;
  // softmax over 4 layer weights (tiny; recompute per thread)
  float w0 = lw[0], w1 = lw[1], w2 = lw[2], w3 = lw[3];
  float mx = fmaxf(fmaxf(w0, w1), fmaxf(w2, w3));
  float e0 = __expf(w0 - mx), e1 = __expf(w1 - mx), e2 = __expf(w2 - mx), e3 = __expf(w3 - mx);
  float inv = 1.0f / (e0 + e1 + e2 + e3);
  e0 *= inv; e1 *= inv; e2 *= inv; e3 *= inv;

  const float4* hs4 = (const float4*)hs;
  const size_t LS4 = (size_t)BS_ * (H_ / 4);         // layer stride in float4
  size_t idx4 = (size_t)row * (H_ / 4) + t;          // each thread: 4 contiguous elems
  float4 a0 = hs4[idx4];
  float4 a1 = hs4[idx4 + LS4];
  float4 a2 = hs4[idx4 + 2 * LS4];
  float4 a3 = hs4[idx4 + 3 * LS4];
  float4 v;
  v.x = e0 * a0.x + e1 * a1.x + e2 * a2.x + e3 * a3.x;
  v.y = e0 * a0.y + e1 * a1.y + e2 * a2.y + e3 * a3.y;
  v.z = e0 * a0.z + e1 * a1.z + e2 * a2.z + e3 * a3.z;
  v.w = e0 * a0.w + e1 * a1.w + e2 * a2.w + e3 * a3.w;

  rs[t] = v.x + v.y + v.z + v.w;
  rq[t] = v.x * v.x + v.y * v.y + v.z * v.z + v.w * v.w;
  __syncthreads();
  for (int o = 128; o > 0; o >>= 1) {
    if (t < o) { rs[t] += rs[t + o]; rq[t] += rq[t + o]; }
    __syncthreads();
  }
  float mean = rs[0] * (1.0f / H_);
  float var  = rq[0] * (1.0f / H_) - mean * mean;
  float rstd = rsqrtf(var + 1e-7f);
  float4 gv = ((const float4*)g)[t];
  float4 bv = ((const float4*)bb)[t];
  float4 y;
  y.x = (v.x - mean) * rstd * gv.x + bv.x;
  y.y = (v.y - mean) * rstd * gv.y + bv.y;
  y.z = (v.z - mean) * rstd * gv.z + bv.z;
  y.w = (v.w - mean) * rstd * gv.w + bv.w;
  ((float4*)xf32)[idx4] = y;
  v4h h16; h16[0] = (_Float16)y.x; h16[1] = (_Float16)y.y;
  h16[2] = (_Float16)y.z; h16[3] = (_Float16)y.w;
  ((v4h*)xf16)[idx4] = h16;
}

// ---- 2) weights f32 -> f16 (Wq|Wk|Wv|Wo concatenated) ----
__global__ void __launch_bounds__(256) k_cvt_w(
    const float* __restrict__ Wq, const float* __restrict__ Wk,
    const float* __restrict__ Wv, const float* __restrict__ Wo,
    _Float16* __restrict__ W16) {
  size_t i4 = (size_t)blockIdx.x * 256 + threadIdx.x;       // float4 index, 0 .. 4*H*H/4-1
  size_t which = i4 / (size_t)(H_ * H_ / 4);
  size_t off = i4 - which * (size_t)(H_ * H_ / 4);
  const float* p = which == 0 ? Wq : which == 1 ? Wk : which == 2 ? Wv : Wo;
  float4 w = ((const float4*)p)[off];
  v4h h16; h16[0] = (_Float16)w.x; h16[1] = (_Float16)w.y;
  h16[2] = (_Float16)w.z; h16[3] = (_Float16)w.w;
  ((v4h*)W16)[i4] = h16;
}

// ---- 3) QKV projections: y = x @ W^T + b. A panel staged to LDS via async copy. ----
__global__ void __launch_bounds__(256) k_gemm_qkv(
    const _Float16* __restrict__ xf16, const _Float16* __restrict__ W16,
    const float* __restrict__ bq, const float* __restrict__ bk, const float* __restrict__ bv,
    _Float16* __restrict__ Qo, _Float16* __restrict__ Ko, _Float16* __restrict__ Vt) {
  __shared__ _Float16 As[16 * H_];            // 32 KB A panel shared by 8 waves
  int t = threadIdx.x;
  int wave = t >> 5, lane = t & 31;
  int r0 = blockIdx.x * 16;
  int c0 = (blockIdx.y * 8 + wave) * 16;
  int z  = blockIdx.z;                        // 0=Q 1=K 2=V

  // Stage 16 rows x 1024 halves = 32KB contiguous, 16B per lane per async op.
  {
    const char* src = (const char*)(xf16 + (size_t)r0 * H_);
    unsigned dst = (unsigned)(unsigned long long)&As[0];
#pragma unroll
    for (int i = 0; i < 8; ++i) {
      int e = t + i * 256;                    // 2048 b128 packets
      async_copy_b128(dst + e * 16, src + e * 16);
    }
    wait_async0();
  }
  __syncthreads();

  const _Float16* W = W16 + (size_t)z * H_ * H_;
  const float* bias = z == 0 ? bq : (z == 1 ? bk : bv);
  const _Float16* arow = As + (size_t)(lane & 15) * H_;
  const _Float16* bcol = W  + (size_t)(c0 + (lane & 15)) * H_;  // row of W == column of W^T
  v8f acc = {};
#pragma unroll 4
  for (int k0 = 0; k0 < H_; k0 += 32) {
    v16h a  = load_a_frag(arow, k0, lane);
    v16h bm = load_b_frag(bcol, k0, lane);
    acc = WMMA_F16(a, bm, acc);
  }
  int n = lane & 15, hi = lane >> 4;
  int h = c0 + n;
  float bsv = bias[h];
  _Float16* out = (z == 0) ? Qo : Ko;
#pragma unroll
  for (int r = 0; r < 8; ++r) {
    int m = r + hi * 8;
    int sg = r0 + m;                          // flattened b*S+s
    _Float16 hv = (_Float16)(acc[r] + bsv);
    if (z < 2) {
      out[(size_t)sg * H_ + h] = hv;
    } else {
      int b = sg >> 11, s = sg & (S_ - 1);
      int head = h >> 9, d = h & (DH_ - 1);
      Vt[((size_t)(b * NH_ + head) * DH_ + d) * S_ + s] = hv;
    }
  }
}

// ---- 4) attention: per (b, head, 16-row q-tile); scores in LDS, 2-pass softmax, P@V ----
__global__ void __launch_bounds__(256) k_attn(
    const _Float16* __restrict__ Q16, const _Float16* __restrict__ K16,
    const _Float16* __restrict__ Vt, const int* __restrict__ mask,
    _Float16* __restrict__ ctx) {
  extern __shared__ char smem[];
  float*    scores = (float*)smem;                            // 16*S f32 = 128 KB
  _Float16* probs  = (_Float16*)(smem + (size_t)16 * S_ * 4); // 16*S f16 =  64 KB
  float*    red    = (float*)(smem + (size_t)16 * S_ * 6);    // 256 f32
  float*    rowmax = red + 256;                               // 16
  float*    rowsum = rowmax + 16;                             // 16

  int qt = blockIdx.x, head = blockIdx.y, b = blockIdx.z;
  int q0 = qt * 16;
  int t = threadIdx.x, wave = t >> 5, lane = t & 31;
  const float scale = 0.04419417382415922f;                   // 1/sqrt(DH)

  // Preload the q-tile A fragments (16 K-steps over DH=512), held in VGPRs.
  const _Float16* qrow = Q16 + (size_t)(b * S_ + q0 + (lane & 15)) * H_ + head * DH_;
  v16h aQ[16];
#pragma unroll
  for (int kk = 0; kk < 16; ++kk) aQ[kk] = load_a_frag(qrow, kk * 32, lane);

  // Phase 1: scores = Q K^T * scale + mask-ext, striped over 8 waves.
  for (int j = wave; j < S_ / 16; j += 8) {
    int sk = j * 16 + (lane & 15);
    const _Float16* kcol = K16 + (size_t)(b * S_ + sk) * H_ + head * DH_;
    v8f acc = {};
#pragma unroll
    for (int kk = 0; kk < 16; ++kk) {
      v16h bm = load_b_frag(kcol, kk * 32, lane);
      acc = WMMA_F16(aQ[kk], bm, acc);
    }
    float ext = (1.0f - (float)mask[b * S_ + sk]) * -10000.0f;
    int hi = lane >> 4;
#pragma unroll
    for (int r = 0; r < 8; ++r) {
      int m = r + hi * 8;
      scores[m * S_ + sk] = acc[r] * scale + ext;
    }
  }
  __syncthreads();

  // Row max (16 rows x 16 threads each), then exp -> unnormalized f16 probs + row sum.
  int row = t >> 4, c0 = t & 15;
  float pm = -3.0e38f;
  for (int c = c0; c < S_; c += 16) pm = fmaxf(pm, scores[row * S_ + c]);
  red[t] = pm;
  __syncthreads();
  if (t < 16) {
    float m = red[t * 16];
#pragma unroll
    for (int i = 1; i < 16; ++i) m = fmaxf(m, red[t * 16 + i]);
    rowmax[t] = m;
  }
  __syncthreads();
  float rm = rowmax[row];
  float ps = 0.f;
  for (int c = c0; c < S_; c += 16) {
    float e = __expf(scores[row * S_ + c] - rm);
    probs[row * S_ + c] = (_Float16)e;
    ps += e;
  }
  red[t] = ps;
  __syncthreads();
  if (t < 16) {
    float s = 0.f;
#pragma unroll
    for (int i = 0; i < 16; ++i) s += red[t * 16 + i];
    rowsum[t] = s;
  }
  __syncthreads();

  // Phase 2: ctx = probs @ V (A from LDS, B from Vt); 1/rowsum folded in epilogue.
  const _Float16* prow = probs + (size_t)(lane & 15) * S_;
  for (int nt = wave; nt < DH_ / 16; nt += 8) {
    int d = nt * 16 + (lane & 15);
    const _Float16* vcol = Vt + ((size_t)(b * NH_ + head) * DH_ + d) * S_;
    v8f acc = {};
    for (int k0 = 0; k0 < S_; k0 += 32) {
      v16h a  = load_a_frag(prow, k0, lane);
      v16h bm = load_b_frag(vcol, k0, lane);
      acc = WMMA_F16(a, bm, acc);
    }
    int hi = lane >> 4;
#pragma unroll
    for (int r = 0; r < 8; ++r) {
      int m = r + hi * 8;
      float val = acc[r] / rowsum[m];
      ctx[(size_t)(b * S_ + q0 + m) * H_ + head * DH_ + d] = (_Float16)val;
    }
  }
}

// ---- 5) output projection + bias + residual -> ypre (f32); A panel via async LDS ----
__global__ void __launch_bounds__(256) k_gemm_out(
    const _Float16* __restrict__ ctx, const _Float16* __restrict__ Wo16,
    const float* __restrict__ bo, const float* __restrict__ xf32,
    float* __restrict__ ypre) {
  __shared__ _Float16 As[16 * H_];            // 32 KB
  int t = threadIdx.x;
  int wave = t >> 5, lane = t & 31;
  int r0 = blockIdx.x * 16;
  int c0 = (blockIdx.y * 8 + wave) * 16;
  {
    const char* src = (const char*)(ctx + (size_t)r0 * H_);
    unsigned dst = (unsigned)(unsigned long long)&As[0];
#pragma unroll
    for (int i = 0; i < 8; ++i) {
      int e = t + i * 256;
      async_copy_b128(dst + e * 16, src + e * 16);
    }
    wait_async0();
  }
  __syncthreads();

  const _Float16* arow = As   + (size_t)(lane & 15) * H_;
  const _Float16* bcol = Wo16 + (size_t)(c0 + (lane & 15)) * H_;
  v8f acc = {};
#pragma unroll 4
  for (int k0 = 0; k0 < H_; k0 += 32) {
    v16h a  = load_a_frag(arow, k0, lane);
    v16h bm = load_b_frag(bcol, k0, lane);
    acc = WMMA_F16(a, bm, acc);
  }
  int n = lane & 15, hi = lane >> 4;
  int h = c0 + n;
  float bsv = bo[h];
#pragma unroll
  for (int r = 0; r < 8; ++r) {
    int m = r + hi * 8;
    size_t idx = (size_t)(r0 + m) * H_ + h;
    ypre[idx] = acc[r] + bsv + xf32[idx];
  }
}

// ---- 6) LN2 in-place (float4 per thread) ----
__global__ void __launch_bounds__(256) k_ln2(
    float* __restrict__ y, const float* __restrict__ g, const float* __restrict__ bb) {
  __shared__ float rs[256], rq[256];
  int row = blockIdx.x;
  int t = threadIdx.x;
  size_t idx4 = (size_t)row * (H_ / 4) + t;
  float4 v = ((const float4*)y)[idx4];
  rs[t] = v.x + v.y + v.z + v.w;
  rq[t] = v.x * v.x + v.y * v.y + v.z * v.z + v.w * v.w;
  __syncthreads();
  for (int o = 128; o > 0; o >>= 1) {
    if (t < o) { rs[t] += rs[t + o]; rq[t] += rq[t + o]; }
    __syncthreads();
  }
  float mean = rs[0] * (1.0f / H_);
  float var  = rq[0] * (1.0f / H_) - mean * mean;
  float rstd = rsqrtf(var + 1e-12f);
  float4 gv = ((const float4*)g)[t];
  float4 bv = ((const float4*)bb)[t];
  float4 o4;
  o4.x = (v.x - mean) * rstd * gv.x + bv.x;
  o4.y = (v.y - mean) * rstd * gv.y + bv.y;
  o4.z = (v.z - mean) * rstd * gv.z + bv.z;
  o4.w = (v.w - mean) * rstd * gv.w + bv.w;
  ((float4*)y)[idx4] = o4;
}

// ---- 7) span means: out[b,n,h] = mean over rows [head, tail) of y (float4/thread) ----
__global__ void __launch_bounds__(256) k_span(
    const float* __restrict__ y, const int* __restrict__ headi, const int* __restrict__ taili,
    float* __restrict__ out) {
  int bs = blockIdx.x;                       // b*NS + span
  int b = bs / NS_;
  int t = threadIdx.x;
  int h0 = headi[bs], t1 = taili[bs];
  if (t1 > S_) t1 = S_;
  float invlen = 1.0f / (float)(t1 - h0);
  const float4* y4 = (const float4*)y;
  float4 s = {0.f, 0.f, 0.f, 0.f};
  for (int r = h0; r < t1; ++r) {
    float4 v = y4[(size_t)(b * S_ + r) * (H_ / 4) + t];
    s.x += v.x; s.y += v.y; s.z += v.z; s.w += v.w;
  }
  s.x *= invlen; s.y *= invlen; s.z *= invlen; s.w *= invlen;
  ((float4*)out)[(size_t)bs * (H_ / 4) + t] = s;
}

extern "C" void kernel_launch(void* const* d_in, const int* in_sizes, int n_in,
                              void* d_out, int out_size, void* d_ws, size_t ws_size,
                              hipStream_t stream) {
  const float* hs   = (const float*)d_in[0];
  const int*   mask = (const int*)d_in[1];
  const int*   shp  = (const int*)d_in[2];
  const int*   stp  = (const int*)d_in[3];
  const float* lw   = (const float*)d_in[4];
  const float* g1   = (const float*)d_in[5];
  const float* b1   = (const float*)d_in[6];
  const float* Wq   = (const float*)d_in[7];
  const float* bq   = (const float*)d_in[8];
  const float* Wk   = (const float*)d_in[9];
  const float* bk   = (const float*)d_in[10];
  const float* Wv   = (const float*)d_in[11];
  const float* bv   = (const float*)d_in[12];
  const float* Wo   = (const float*)d_in[13];
  const float* bo   = (const float*)d_in[14];
  const float* g2   = (const float*)d_in[15];
  const float* b2   = (const float*)d_in[16];
  (void)in_sizes; (void)n_in; (void)out_size; (void)ws_size;

  // Workspace layout (~296 MB)
  char* ws = (char*)d_ws;
  size_t off = 0;
  float*    xf32  = (float*)(ws + off);    off += (size_t)BS_ * H_ * 4;   // 64 MB
  _Float16* xf16  = (_Float16*)(ws + off); off += (size_t)BS_ * H_ * 2;   // 32 MB
  _Float16* W16   = (_Float16*)(ws + off); off += (size_t)4 * H_ * H_ * 2;//  8 MB
  _Float16* Q16   = (_Float16*)(ws + off); off += (size_t)BS_ * H_ * 2;   // 32 MB
  _Float16* K16   = (_Float16*)(ws + off); off += (size_t)BS_ * H_ * 2;   // 32 MB
  _Float16* Vt16  = (_Float16*)(ws + off); off += (size_t)BS_ * H_ * 2;   // 32 MB
  _Float16* ctx16 = (_Float16*)(ws + off); off += (size_t)BS_ * H_ * 2;   // 32 MB
  float*    ypre  = (float*)(ws + off);    off += (size_t)BS_ * H_ * 4;   // 64 MB

  k_fuse_ln1<<<BS_, 256, 0, stream>>>(hs, lw, g1, b1, xf32, xf16);
  k_cvt_w<<<(4 * H_ * H_ / 4) / 256, 256, 0, stream>>>(Wq, Wk, Wv, Wo, W16);
  k_gemm_qkv<<<dim3(BS_ / 16, H_ / 128, 3), 256, 0, stream>>>(xf16, W16, bq, bk, bv, Q16, K16, Vt16);

  size_t smem = (size_t)16 * S_ * 4 + (size_t)16 * S_ * 2 + 256 * 4 + 32 * 4;  // ~193 KB (<= 320 KB/WGP)
  k_attn<<<dim3(S_ / 16, NH_, B_), 256, smem, stream>>>(Q16, K16, Vt16, mask, ctx16);

  k_gemm_out<<<dim3(BS_ / 16, H_ / 128), 256, 0, stream>>>(ctx16, W16 + (size_t)3 * H_ * H_, bo, xf32, ypre);
  k_ln2<<<BS_, 256, 0, stream>>>(ypre, g2, b2);
  k_span<<<B_ * NS_, 256, 0, stream>>>(ypre, shp, stp, (float*)d_out);
}
